// DFFusionBlockMask_11879879544369
// MI455X (gfx1250) — compile-verified, matrix operand order checked by reading, not run on hardware
//
#include <hip/hip_runtime.h>
#include <stdint.h>

// ---------------------------------------------------------------------------
// MI455X (gfx1250) fused windowed attention.
//   DIM=256, DH=32, HEADS=8, AG=6, WS=7 -> N=294 tokens/window, 256 windows.
// bf16 WMMA everywhere (v_wmma_f32_16x16x32_bf16); dh=32 == WMMA K exactly.
// Fragment loads are 2 x b128 per fragment; all global addressing is
// uniform-base + unsigned 32-bit offset to enable SADDR-form loads.
// ---------------------------------------------------------------------------

typedef __bf16 bf16_t;
typedef __attribute__((ext_vector_type(16))) __bf16 v16bf;
typedef __attribute__((ext_vector_type(8)))  __bf16 v8bf;
typedef __attribute__((ext_vector_type(8)))  float  v8f;
typedef __attribute__((ext_vector_type(4)))  float  v4f;

#define DIMX   256
#define DH     32
#define HEADS  8
#define AGENTS 6
#define WSZ    7
#define NTOK   294          // AGENTS*WSZ*WSZ
#define MP     304          // 19*16 padded query rows
#define KP     320          // 10*32 padded key rows
#define NWIN   256          // 16*16 windows
#define MTILES 19

static __device__ __forceinline__ v8f vzero8() {
  v8f z = {0.f, 0.f, 0.f, 0.f, 0.f, 0.f, 0.f, 0.f};
  return z;
}

static __device__ __forceinline__ v16bf frag_zero() {
  v16bf f;
#pragma unroll
  for (int i = 0; i < 16; ++i) f[i] = (bf16_t)0.0f;
  return f;
}

// ISA 7.12.2 16-bit A layout (16x32): lane L (m = L&15, khalf = L>>4) holds
// K = {8*khalf .. 8*khalf+7} then {16+8*khalf .. +7}: two contiguous 16-byte
// chunks -> two b128 loads.  B fragments mirror this with the lane indexing N.
static __device__ __forceinline__ v16bf frag_ld_bf16(const bf16_t* base,
                                                     unsigned off, int khalf) {
  const bf16_t* p = base + off + 8 * khalf;
  const v8bf a = *(const v8bf*)p;
  const v8bf b = *(const v8bf*)(p + 16);
  return __builtin_shufflevector(a, b, 0, 1, 2, 3, 4, 5, 6, 7, 8, 9, 10, 11,
                                 12, 13, 14, 15);
}

static __device__ __forceinline__ v16bf frag_ld_f32(const float* base,
                                                    unsigned off, int khalf) {
  const float* p = base + off + 8 * khalf;
  const v4f a0 = *(const v4f*)p;
  const v4f a1 = *(const v4f*)(p + 4);
  const v4f b0 = *(const v4f*)(p + 16);
  const v4f b1 = *(const v4f*)(p + 20);
  v16bf f;
#pragma unroll
  for (int i = 0; i < 4; ++i) {
    f[i]      = (bf16_t)a0[i];
    f[4 + i]  = (bf16_t)a1[i];
    f[8 + i]  = (bf16_t)b0[i];
    f[12 + i] = (bf16_t)b1[i];
  }
  return f;
}

static __device__ __forceinline__ v8f wmma_bf16(v16bf a, v16bf b, v8f c) {
  return __builtin_amdgcn_wmma_f32_16x16x32_bf16(false, a, false, b, (short)0, c,
                                                 false, false);
}

// ---------------------------------------------------------------------------
// Kernel 1: weight transpose+cast (q-scale folded) and dense bias table.
// ---------------------------------------------------------------------------
__global__ __launch_bounds__(256) void prep_kernel(
    const float* __restrict__ wqkv, const float* __restrict__ wout,
    const float* __restrict__ btab, const int* __restrict__ relidx,
    bf16_t* __restrict__ wqkvT, bf16_t* __restrict__ woutT,
    float* __restrict__ bias8) {
  const int tid = blockIdx.x * blockDim.x + threadIdx.x;
  const int nth = gridDim.x * blockDim.x;

  // wqkvT[j][k] = wqkv[k][j] * (j<256 ? dh^-0.5 : 1)   (768 x 256, bf16)
  for (int i = tid; i < 768 * DIMX; i += nth) {
    const int j = i >> 8, k = i & 255;
    const float s = (j < DIMX) ? 0.17677669529663687f : 1.0f;
    wqkvT[i] = (bf16_t)(wqkv[k * 768 + j] * s);
  }
  // woutT[j][k] = wout[k][j]   (256 x 256, bf16)
  for (int i = tid; i < DIMX * DIMX; i += nth) {
    const int j = i >> 8, k = i & 255;
    woutT[i] = (bf16_t)wout[k * DIMX + j];
  }
  // bias8[h][i][j] dense, padded (8, 304, 320) f32 -> stays L2 resident
  for (int i = tid; i < HEADS * MP * KP; i += nth) {
    const int h = i / (MP * KP);
    const int rem = i % (MP * KP);
    const int r = rem / KP, c = rem % KP;
    float v = 0.0f;
    if (r < NTOK && c < NTOK) v = btab[relidx[r * NTOK + c] * HEADS + h];
    bias8[i] = v;
  }
}

// ---------------------------------------------------------------------------
// Kernel 2: per (window, head) fused QKV projection + flash attention.
// grid = (HEADS, NWIN), block = 256 threads = 8 waves.
// ---------------------------------------------------------------------------
__global__ __launch_bounds__(256) void attn_kernel(
    const float* __restrict__ x, const int* __restrict__ mask,
    const float* __restrict__ bias8, const bf16_t* __restrict__ wqkvT,
    bf16_t* __restrict__ attn_out) {
  __shared__ bf16_t sK[KP * DH];        // keys, row-major (token, dh)
  __shared__ bf16_t sVt[DH * KP];       // values transposed (dh, token)
  __shared__ float  sMaskAdd[KP];       // 0 or -1e30 per key token
  __shared__ bf16_t sP[8 * 512];        // per-wave 16x32 C->A relayout scratch

  const int tid  = threadIdx.x;
  const int wave = tid >> 5;
  const int lane = tid & 31;
  const int lo   = lane & 15;           // N / M index within tile
  const int hi   = lane >> 4;           // khalf
  const int head = blockIdx.x;
  const int win  = blockIdx.y;
  const int gx = win >> 4, gy = win & 15;

  // ---- phase 0: key mask addend + zero the padded K/V tails --------------
  for (int j = tid; j < KP; j += 256) {
    float madd = -1e30f;
    if (j < NTOK) {
      const int l = j / 49, rr = j % 49, w1 = rr / 7, w2 = rr % 7;
      const int midx = (((gx * 16 + gy) * WSZ + w1) * WSZ + w2) * AGENTS + l;
      madd = mask[midx] ? 0.0f : -1e30f;
    }
    sMaskAdd[j] = madd;
  }
  for (int i = tid; i < (KP - MP) * DH; i += 256) sK[MP * DH + i] = (bf16_t)0.0f;
  for (int i = tid; i < DH * (KP - MP); i += 256) {
    const int n = i >> 4, jj = MP + (i & 15);
    sVt[n * KP + jj] = (bf16_t)0.0f;
  }

  // ---- phase 1: Q/K/V head-slice projection via WMMA ---------------------
  // Each wave owns query tiles mt = wave, wave+8, wave+16 for BOTH phases,
  // so Q fragments can live in registers (after an LDS relayout bounce).
  v16bf qfrag[3];
#pragma unroll
  for (int c = 0; c < 3; ++c) qfrag[c] = frag_zero();

#pragma unroll
  for (int c = 0; c < 3; ++c) {
    const int mt = wave + 8 * c;
    if (mt < MTILES) {
      const int m = mt * 16 + lo;       // token row this lane feeds into A
      const bool valid = (m < NTOK);
      unsigned xoff = 0;
      if (valid) {
        const int l = m / 49, rr = m % 49, w1 = rr / 7, w2 = rr % 7;
        xoff = (unsigned)((((l * 16 + gx) * 16 + gy) * WSZ + w1) * WSZ + w2) * DIMX;
      }
      v8f acc[6];
#pragma unroll
      for (int i = 0; i < 6; ++i) acc[i] = vzero8();
#pragma unroll
      for (int kt = 0; kt < 8; ++kt) {
        const v16bf a = valid ? frag_ld_f32(x, xoff + kt * 32, hi) : frag_zero();
#pragma unroll
        for (int mat = 0; mat < 3; ++mat) {
#pragma unroll
          for (int nt = 0; nt < 2; ++nt) {
            const unsigned woff =
                (unsigned)(mat * DIMX + head * DH + nt * 16 + lo) * DIMX + kt * 32;
            const v16bf b = frag_ld_bf16(wqkvT, woff, hi);
            acc[mat * 2 + nt] = wmma_bf16(a, b, acc[mat * 2 + nt]);
          }
        }
      }
      // scatter results: Q -> per-wave scratch (A layout source), K row-major,
      // V transposed (so V B-fragments load contiguously along keys).
      bf16_t* pw = sP + wave * 512;
#pragma unroll
      for (int r = 0; r < 8; ++r) {
        const int row = mt * 16 + r + 8 * hi;
        pw[(r + 8 * hi) * 32 + lo]      = (bf16_t)acc[0][r];
        pw[(r + 8 * hi) * 32 + 16 + lo] = (bf16_t)acc[1][r];
        sK[row * DH + lo]       = (bf16_t)acc[2][r];
        sK[row * DH + 16 + lo]  = (bf16_t)acc[3][r];
        sVt[lo * KP + row]        = (bf16_t)acc[4][r];
        sVt[(16 + lo) * KP + row] = (bf16_t)acc[5][r];
      }
      qfrag[c] = frag_ld_bf16(pw, (unsigned)(lo * 32), hi);  // same-wave DS order
    }
  }
  __syncthreads();

  // ---- phase 2: flash attention over 10 key chunks of 32 -----------------
#pragma unroll
  for (int c = 0; c < 3; ++c) {
    const int mt = wave + 8 * c;
    if (mt < MTILES) {
      const v16bf qa = qfrag[c];
      const unsigned boff = (unsigned)(head * MP + mt * 16 + 8 * hi) * KP;
      float mrow[8], lrow[8];
#pragma unroll
      for (int r = 0; r < 8; ++r) { mrow[r] = -1e30f; lrow[r] = 0.0f; }
      v8f o0 = vzero8(), o1 = vzero8();

      for (int jc = 0; jc < 10; ++jc) {
        const int j0 = jc * 32;
        if (jc < 9) __builtin_prefetch(&bias8[boff + j0 + 32], 0, 1);
        const v16bf kb0 = frag_ld_bf16(sK, (unsigned)((j0 + lo) * DH), hi);
        const v16bf kb1 = frag_ld_bf16(sK, (unsigned)((j0 + 16 + lo) * DH), hi);
        v8f s0 = wmma_bf16(qa, kb0, vzero8());
        v8f s1 = wmma_bf16(qa, kb1, vzero8());
        const float ma0 = sMaskAdd[j0 + lo];
        const float ma1 = sMaskAdd[j0 + 16 + lo];
        float e0[8], e1[8], cm[8];
#pragma unroll
        for (int r = 0; r < 8; ++r) {
          e0[r] = s0[r] + bias8[boff + r * KP + j0 + lo] + ma0;
          e1[r] = s1[r] + bias8[boff + r * KP + j0 + 16 + lo] + ma1;
          cm[r] = fmaxf(e0[r], e1[r]);
        }
        // row max across the 16-lane N group (C layout: lanes hold columns)
#pragma unroll
        for (int d = 1; d < 16; d <<= 1) {
#pragma unroll
          for (int r = 0; r < 8; ++r) cm[r] = fmaxf(cm[r], __shfl_xor(cm[r], d, 32));
        }
        float ps[8];
#pragma unroll
        for (int r = 0; r < 8; ++r) {
          const float nm = fmaxf(mrow[r], cm[r]);
          const float sc = __expf(mrow[r] - nm);
          mrow[r] = nm;
          e0[r] = __expf(e0[r] - nm);
          e1[r] = __expf(e1[r] - nm);
          ps[r] = e0[r] + e1[r];
          lrow[r] *= sc;
          o0[r] *= sc;
          o1[r] *= sc;
        }
#pragma unroll
        for (int d = 1; d < 16; d <<= 1) {
#pragma unroll
          for (int r = 0; r < 8; ++r) ps[r] += __shfl_xor(ps[r], d, 32);
        }
#pragma unroll
        for (int r = 0; r < 8; ++r) lrow[r] += ps[r];

        // P tile C->A relayout through per-wave scratch (in-order DS)
        bf16_t* pw = sP + wave * 512;
#pragma unroll
        for (int r = 0; r < 8; ++r) {
          pw[(r + 8 * hi) * 32 + lo]      = (bf16_t)e0[r];
          pw[(r + 8 * hi) * 32 + 16 + lo] = (bf16_t)e1[r];
        }
        const v16bf pa  = frag_ld_bf16(pw, (unsigned)(lo * 32), hi);
        const v16bf vb0 = frag_ld_bf16(sVt, (unsigned)(lo * KP + j0), hi);
        const v16bf vb1 = frag_ld_bf16(sVt, (unsigned)((16 + lo) * KP + j0), hi);
        o0 = wmma_bf16(pa, vb0, o0);
        o1 = wmma_bf16(pa, vb1, o1);
      }

      // normalize + store head slice (bf16, (win, 304, 256) layout)
#pragma unroll
      for (int r = 0; r < 8; ++r) {
        const float inv = 1.0f / fmaxf(lrow[r], 1e-30f);
        const int row = mt * 16 + r + 8 * hi;
        const float v0 = (row < NTOK) ? o0[r] * inv : 0.0f;
        const float v1 = (row < NTOK) ? o1[r] * inv : 0.0f;
        const unsigned doff =
            (unsigned)(win * MP + row) * DIMX + (unsigned)(head * DH);
        attn_out[doff + lo]      = (bf16_t)v0;
        attn_out[doff + 16 + lo] = (bf16_t)v1;
      }
    }
  }
}

// ---------------------------------------------------------------------------
// Kernel 3: output projection + scatter to (b,l,gx,gy,w1,w2,d) layout.
// grid = (19 row tiles, 256 windows), block = 256 threads.
// ---------------------------------------------------------------------------
__global__ __launch_bounds__(256) void outproj_kernel(
    const bf16_t* __restrict__ attn, const bf16_t* __restrict__ woutT,
    float* __restrict__ out) {
  __shared__ bf16_t sA[16 * DIMX];      // 8 KB staged A tile
  const int tid = threadIdx.x;
  const int wave = tid >> 5, lane = tid & 31;
  const int lo = lane & 15, hi = lane >> 4;
  const int mt = blockIdx.x;
  const int win = blockIdx.y;
  const int gx = win >> 4, gy = win & 15;

  {
    const uint4* src =
        (const uint4*)(attn + (size_t)(win * MP + mt * 16) * DIMX);
    uint4* dst = (uint4*)sA;
    for (int i = tid; i < 16 * DIMX * 2 / 16; i += 256) dst[i] = src[i];
  }
  __syncthreads();

  const int nt0 = wave * 2;             // 8 waves x 2 N-tiles = 16 tiles of 16
  v8f acc0 = vzero8(), acc1 = vzero8();
#pragma unroll
  for (int kt = 0; kt < 8; ++kt) {
    const v16bf a  = frag_ld_bf16(sA, (unsigned)(lo * DIMX + kt * 32), hi);
    const v16bf b0 = frag_ld_bf16(
        woutT, (unsigned)((nt0 * 16 + lo) * DIMX + kt * 32), hi);
    const v16bf b1 = frag_ld_bf16(
        woutT, (unsigned)(((nt0 + 1) * 16 + lo) * DIMX + kt * 32), hi);
    acc0 = wmma_bf16(a, b0, acc0);
    acc1 = wmma_bf16(a, b1, acc1);
  }
#pragma unroll
  for (int r = 0; r < 8; ++r) {
    const int row = mt * 16 + r + 8 * hi;
    if (row < NTOK) {
      const int l = row / 49, rr = row % 49, w1 = rr / 7, w2 = rr % 7;
      const unsigned ooff =
          (unsigned)((((l * 16 + gx) * 16 + gy) * WSZ + w1) * WSZ + w2) * DIMX;
      out[ooff + nt0 * 16 + lo]       = acc0[r];
      out[ooff + (nt0 + 1) * 16 + lo] = acc1[r];
    }
  }
}

// ---------------------------------------------------------------------------
extern "C" void kernel_launch(void* const* d_in, const int* in_sizes, int n_in,
                              void* d_out, int out_size, void* d_ws, size_t ws_size,
                              hipStream_t stream) {
  (void)in_sizes; (void)n_in; (void)out_size; (void)ws_size;
  const float* x      = (const float*)d_in[0];
  const int*   mask   = (const int*)d_in[1];
  const float* wqkv   = (const float*)d_in[2];
  const float* wout   = (const float*)d_in[3];
  const float* btab   = (const float*)d_in[4];
  const int*   relidx = (const int*)d_in[5];

  char* ws = (char*)d_ws;
  const size_t off_woutT = (size_t)768 * DIMX * 2;               // 393216
  const size_t off_bias  = off_woutT + (size_t)DIMX * DIMX * 2;  // 524288
  const size_t off_attn  = off_bias + (size_t)HEADS * MP * KP * 4;  // 3637248
  bf16_t* wqkvT   = (bf16_t*)ws;
  bf16_t* woutT   = (bf16_t*)(ws + off_woutT);
  float*  bias8   = (float*)(ws + off_bias);
  bf16_t* attnbuf = (bf16_t*)(ws + off_attn);   // (256, 304, 256) bf16 ~40MB

  prep_kernel<<<dim3(768), dim3(256), 0, stream>>>(wqkv, wout, btab, relidx,
                                                   wqkvT, woutT, bias8);
  attn_kernel<<<dim3(HEADS, NWIN), dim3(256), 0, stream>>>(x, mask, bias8,
                                                           wqkvT, attnbuf);
  outproj_kernel<<<dim3(MTILES, NWIN), dim3(256), 0, stream>>>(attnbuf, woutT,
                                                               (float*)d_out);
}
